// global_attn_44899588112542
// MI455X (gfx1250) — matrix operand order repaired and checked by reading, not compiled, not run on hardware
//
#include <hip/hip_runtime.h>
#include <hip/hip_bf16.h>

// ---------------- problem constants ----------------
#define BG   8
#define PP   2048
#define KK   20
#define NN   (BG * PP)        // 16384 points
#define MM   (BG * KK)        // 160 fps nodes
#define CIN  64
#define CC   128
#define EE   (NN * KK)        // 327680 edges
#define EPSV 1e-5f

typedef __attribute__((ext_vector_type(16))) _Float16 v16h;
typedef __attribute__((ext_vector_type(8)))  _Float16 h8;
typedef __attribute__((ext_vector_type(8)))  float    v8f;

// ---------------- tiny utility kernels ----------------
__global__ void k_zero(float* __restrict__ p, long n) {
    long i = (long)blockIdx.x * blockDim.x + threadIdx.x;
    if (i < n) p[i] = 0.f;
}

__global__ void k_cvt_f32_to_f16(const float* __restrict__ s, _Float16* __restrict__ d, long n) {
    long i = (long)blockIdx.x * blockDim.x + threadIdx.x;
    if (i < n) d[i] = (_Float16)s[i];
}

// ---------------- stage 1: knn(k=1) vs original fps positions ----------------
__global__ void k_knn(const float* __restrict__ pos, const int* __restrict__ fps_idx,
                      int* __restrict__ e1, float* __restrict__ ek1) {
    int n = blockIdx.x * blockDim.x + threadIdx.x;
    if (n >= NN) return;
    int b = n / PP;
    float px = pos[n * 3 + 0], py = pos[n * 3 + 1], pz = pos[n * 3 + 2];
    float bd = 3.4e38f; int bi = 0;
    for (int k = 0; k < KK; ++k) {
        int m = b * KK + k;
        int src = fps_idx[m];
        float dx = px - pos[src * 3 + 0];
        float dy = py - pos[src * 3 + 1];
        float dz = pz - pos[src * 3 + 2];
        float d2 = dx * dx + dy * dy + dz * dz;
        if (d2 < bd) { bd = d2; bi = k; }
    }
    e1[n]  = b * KK + bi;
    ek1[n] = 1.f / (1.f + 5.f * bd);
}

// ---------------- stage 2: kernel-weighted scatter-mean onto fps nodes ----------------
__global__ void k_scatter(const float* __restrict__ x, const float* __restrict__ pos,
                          const int* __restrict__ e1, const float* __restrict__ ek1,
                          float* __restrict__ accX, float* __restrict__ accP,
                          float* __restrict__ cnt) {
    long i = (long)blockIdx.x * blockDim.x + threadIdx.x;
    if (i >= (long)NN * CIN) return;
    int n = (int)(i >> 6);
    int c = (int)(i & 63);
    int m = e1[n];
    float w = ek1[n];
    atomicAdd(&accX[m * CIN + c], w * x[(long)n * CIN + c]);
    if (c < 3) atomicAdd(&accP[m * 3 + c], w * pos[n * 3 + c]);
    if (c == 0) atomicAdd(&cnt[m], 1.f);
}

__global__ void k_fps_final(const float* __restrict__ accX, const float* __restrict__ accP,
                            const float* __restrict__ cnt,
                            float* __restrict__ fps_x, float* __restrict__ fps_pos) {
    int i = blockIdx.x * blockDim.x + threadIdx.x;
    if (i >= MM * CIN) return;
    int m = i >> 6, c = i & 63;
    float inv = 1.f / fmaxf(cnt[m], 1.f);
    fps_x[m * CIN + c] = accX[m * CIN + c] * inv;
    if (c < 3) fps_pos[m * 3 + c] = accP[m * 3 + c] * inv;
}

// ---------------- stage 3: k/v projections (M=160 rows -> scalar) ----------------
__global__ void k_kv(const float* __restrict__ fps_x,
                     const float* __restrict__ Wk, const float* __restrict__ bk,
                     const float* __restrict__ Wv, const float* __restrict__ bv,
                     float* __restrict__ xk, float* __restrict__ xv) {
    int i = blockIdx.x * blockDim.x + threadIdx.x;
    if (i >= MM * CC) return;
    int m = i >> 7, o = i & 127;
    float sk = bk[o], sv = bv[o];
    for (int c = 0; c < CIN; ++c) {
        float f = fps_x[m * CIN + c];
        sk += f * Wk[o * CIN + c];
        sv += f * Wv[o * CIN + c];
    }
    xk[i] = sk; xv[i] = sv;
}

// ---------------- WMMA GEMM: Y[rows,128] = Ah[rows,Kd] @ W[128,Kd]^T + bias ----------------
// 128 threads = 4 waves; wave w -> 16-row tile; 8 column tiles of 16 -> full 128 cols.
__global__ __launch_bounds__(128) void k_gemm_wmma(const _Float16* __restrict__ A,
                                                   const _Float16* __restrict__ W,
                                                   const float* __restrict__ bias,
                                                   float* __restrict__ Y, int Kd) {
    __shared__ _Float16 Wl[128 * 136];          // padded: ldw = Kd + 8 halfs
    const int ldw = Kd + 8;
    {   // cooperative stage of the whole [128,Kd] weight into LDS
        int c = threadIdx.x;                    // one output-channel row per thread
        for (int k = 0; k < Kd; k += 8)
            *(h8*)&Wl[c * ldw + k] = *(const h8*)&W[c * Kd + k];
    }
    __syncthreads();

    const int wave = threadIdx.x >> 5;
    const int lane = threadIdx.x & 31;
    const int r16  = lane & 15;                 // A row within tile / B-D column
    const int hs   = lane >> 4;                 // half-select
    const long rowBase = (long)blockIdx.x * 64 + wave * 16;

    v8f acc[8] = {};
    const _Float16* arow = A + (rowBase + r16) * Kd;

    for (int k0 = 0; k0 < Kd; k0 += 32) {
        // A fragment (16x32 f16): lanes 0-15 K={0-7,16-23}, lanes 16-31 K={8-15,24-31}
        h8 alo = *(const h8*)(arow + k0 + hs * 8);
        h8 ahi = *(const h8*)(arow + k0 + hs * 8 + 16);
        v16h af = __builtin_shufflevector(alo, ahi, 0,1,2,3,4,5,6,7,8,9,10,11,12,13,14,15);
#pragma unroll
        for (int nt = 0; nt < 8; ++nt) {
            // B fragment (32x16): lane column = r16, lanes 0-15 K={0-7,16-23}, 16-31 K={8-15,24-31}
            const _Float16* wrow = &Wl[(nt * 16 + r16) * ldw + k0 + hs * 8];
            h8 blo = *(const h8*)(wrow);
            h8 bhi = *(const h8*)(wrow + 16);
            v16h bf = __builtin_shufflevector(blo, bhi, 0,1,2,3,4,5,6,7,8,9,10,11,12,13,14,15);
            acc[nt] = __builtin_amdgcn_wmma_f32_16x16x32_f16(
                false, af, false, bf, (short)0, acc[nt], false, false);
        }
    }

    // D layout: lane col = r16; VGPR r -> row r + 8*hs
    const long orow = rowBase + hs * 8;
#pragma unroll
    for (int nt = 0; nt < 8; ++nt) {
        int col = nt * 16 + r16;
        float bv = bias[col];
#pragma unroll
        for (int r = 0; r < 8; ++r)
            Y[(orow + r) * CC + col] = acc[nt][r] + bv;
    }
}

// ---------------- pos-MLP layer 1 (Kd=3 -> scalar, coalesced over channels) ----------------
__global__ void k_posmlp1(const float* __restrict__ pos, const float* __restrict__ fps_pos,
                          const float* __restrict__ pW1, const float* __restrict__ pb1,
                          float* __restrict__ Y) {
    long i = (long)blockIdx.x * blockDim.x + threadIdx.x;
    if (i >= (long)EE * CC) return;
    long e = i >> 7; int c = (int)(i & 127);
    int n = (int)(e / KK), kk = (int)(e % KK);
    int m = (n / PP) * KK + kk;
    float rx = pos[n * 3 + 0] - fps_pos[m * 3 + 0];
    float ry = pos[n * 3 + 1] - fps_pos[m * 3 + 1];
    float rz = pos[n * 3 + 2] - fps_pos[m * 3 + 2];
    Y[i] = rx * pW1[c * 3 + 0] + ry * pW1[c * 3 + 1] + rz * pW1[c * 3 + 2] + pb1[c];
}

// ---------------- BN batch stats: per-channel sum & sumsq over E rows ----------------
__global__ __launch_bounds__(256) void k_stats(const float* __restrict__ Y,
                                               float* __restrict__ sum, float* __restrict__ sq) {
    __shared__ float ssum[256], ssq[256];
    const int t = threadIdx.x, c = t & 127, h = t >> 7;
    const int rowsPerBlk = EE / 256;            // 1280
    const long base = (long)blockIdx.x * rowsPerBlk;
    float s = 0.f, q = 0.f;
    for (int r = h; r < rowsPerBlk; r += 2) {
        float v = Y[(base + r) * CC + c];
        s += v; q += v * v;
    }
    ssum[t] = s; ssq[t] = q;
    __syncthreads();
    if (t < 128) {
        atomicAdd(&sum[c], ssum[t] + ssum[t + 128]);
        atomicAdd(&sq[c],  ssq[t]  + ssq[t + 128]);
    }
}

__global__ void k_bn_finalize(const float* __restrict__ sum, const float* __restrict__ sq,
                              const float* __restrict__ g, const float* __restrict__ be,
                              float* __restrict__ scale, float* __restrict__ shift) {
    int c = threadIdx.x;
    if (c >= CC) return;
    float mean = sum[c] / (float)EE;
    float var  = sq[c] / (float)EE - mean * mean;   // biased, matches jnp.var
    float sc   = g[c] * rsqrtf(var + EPSV);
    scale[c] = sc;
    shift[c] = be[c] - mean * sc;
}

// ---------------- BN+ReLU variants ----------------
__global__ void k_bnrelu_h(const float* __restrict__ Y, const float* __restrict__ scale,
                           const float* __restrict__ shift, _Float16* __restrict__ D) {
    long i = (long)blockIdx.x * blockDim.x + threadIdx.x;
    if (i >= (long)EE * CC) return;
    int c = (int)(i & 127);
    D[i] = (_Float16)fmaxf(0.f, Y[i] * scale[c] + shift[c]);
}

// delta = ReLU(BN(Y)); attn-MLP input = f16(delta + x_q[n] - x_k[m])
__global__ void k_delta_attnin(const float* __restrict__ Y, const float* __restrict__ scale,
                               const float* __restrict__ shift, const float* __restrict__ xq,
                               const float* __restrict__ xk, float* __restrict__ delta,
                               _Float16* __restrict__ D) {
    long i = (long)blockIdx.x * blockDim.x + threadIdx.x;
    if (i >= (long)EE * CC) return;
    long e = i >> 7; int c = (int)(i & 127);
    int n = (int)(e / KK), kk = (int)(e % KK);
    int m = (n / PP) * KK + kk;
    float d = fmaxf(0.f, Y[i] * scale[c] + shift[c]);
    delta[i] = d;
    D[i] = (_Float16)(d + xq[(long)n * CC + c] - xk[m * CC + c]);
}

__global__ void k_bnrelu_inplace(float* __restrict__ Y, const float* __restrict__ scale,
                                 const float* __restrict__ shift) {
    long i = (long)blockIdx.x * blockDim.x + threadIdx.x;
    if (i >= (long)EE * CC) return;
    int c = (int)(i & 127);
    Y[i] = fmaxf(0.f, Y[i] * scale[c] + shift[c]);
}

// ---------------- softmax over K + weighted (v + delta) reduction ----------------
__global__ void k_combine(const float* __restrict__ alpha, const float* __restrict__ delta,
                          const float* __restrict__ xv, float* __restrict__ out) {
    long i = (long)blockIdx.x * blockDim.x + threadIdx.x;
    if (i >= (long)NN * CC) return;
    int n = (int)(i >> 7), c = (int)(i & 127);
    int b = n / PP;
    float a[KK];
    float mx = -3.4e38f;
    for (int k = 0; k < KK; ++k) {
        a[k] = alpha[((long)n * KK + k) * CC + c];
        mx = fmaxf(mx, a[k]);
    }
    float den = 0.f;
    for (int k = 0; k < KK; ++k) { a[k] = __expf(a[k] - mx); den += a[k]; }
    float inv = 1.f / den;
    float acc = 0.f;
    for (int k = 0; k < KK; ++k) {
        long e = (long)n * KK + k;
        int m = b * KK + k;
        acc += a[k] * inv * (xv[m * CC + c] + delta[e * CC + c]);
    }
    out[i] = acc;
}

// ---------------- host driver ----------------
extern "C" void kernel_launch(void* const* d_in, const int* in_sizes, int n_in,
                              void* d_out, int out_size, void* d_ws, size_t ws_size,
                              hipStream_t stream) {
    (void)in_sizes; (void)n_in; (void)out_size; (void)ws_size;
    const float* x    = (const float*)d_in[0];
    const float* pos  = (const float*)d_in[1];
    const float* Wq   = (const float*)d_in[2];  const float* bq  = (const float*)d_in[3];
    const float* Wk   = (const float*)d_in[4];  const float* bk  = (const float*)d_in[5];
    const float* Wv   = (const float*)d_in[6];  const float* bv  = (const float*)d_in[7];
    const float* pW1  = (const float*)d_in[8];  const float* pb1 = (const float*)d_in[9];
    const float* pg1  = (const float*)d_in[10]; const float* pbe1= (const float*)d_in[11];
    const float* pW2  = (const float*)d_in[12]; const float* pb2 = (const float*)d_in[13];
    const float* pg2  = (const float*)d_in[14]; const float* pbe2= (const float*)d_in[15];
    const float* aW1  = (const float*)d_in[16]; const float* ab1 = (const float*)d_in[17];
    const float* ag1  = (const float*)d_in[18]; const float* abe1= (const float*)d_in[19];
    const float* aW2  = (const float*)d_in[20]; const float* ab2 = (const float*)d_in[21];
    const float* ag2  = (const float*)d_in[22]; const float* abe2= (const float*)d_in[23];
    const int*   fps  = (const int*)d_in[24];
    float* out = (float*)d_out;

    // workspace carve-up (256B aligned)
    char* base = (char*)d_ws;
    size_t off = 0;
    auto carve = [&](size_t bytes) -> char* {
        char* p = base + off;
        off = (off + bytes + 255) & ~(size_t)255;
        return p;
    };
    float*    Y      = (float*)   carve((size_t)EE * CC * 4);
    float*    delta  = (float*)   carve((size_t)EE * CC * 4);
    _Float16* AinH   = (_Float16*)carve((size_t)EE * CC * 2);
    _Float16* xh     = (_Float16*)carve((size_t)NN * CIN * 2);
    _Float16* WqH    = (_Float16*)carve((size_t)CC * CIN * 2);
    _Float16* pW2H   = (_Float16*)carve((size_t)CC * CC * 2);
    _Float16* aW1H   = (_Float16*)carve((size_t)CC * CC * 2);
    _Float16* aW2H   = (_Float16*)carve((size_t)CC * CC * 2);
    float*    xq     = (float*)   carve((size_t)NN * CC * 4);
    float*    xk     = (float*)   carve((size_t)MM * CC * 4);
    float*    xv     = (float*)   carve((size_t)MM * CC * 4);
    float*    fpsx   = (float*)   carve((size_t)MM * CIN * 4);
    float*    fpsp   = (float*)   carve((size_t)MM * 3 * 4);
    float*    accX   = (float*)   carve((size_t)(MM * CIN + MM * 3 + MM) * 4); // contiguous acc block
    float*    accP   = accX + MM * CIN;
    float*    cnt    = accP + MM * 3;
    int*      e1     = (int*)     carve((size_t)NN * 4);
    float*    ek1    = (float*)   carve((size_t)NN * 4);
    float*    stSum  = (float*)   carve(128 * 4);
    float*    stSq   = (float*)   carve(128 * 4);
    float*    stScale= (float*)   carve(128 * 4);
    float*    stShift= (float*)   carve(128 * 4);

    const long ECtot = (long)EE * CC;
    dim3 blk256(256);
    auto g = [](long n) { return dim3((unsigned)((n + 255) / 256)); };

    // ---- stage 0: zero accumulators, knn, scatter-mean, k/v ----
    k_zero<<<g(MM * CIN + MM * 3 + MM), blk256, 0, stream>>>(accX, MM * CIN + MM * 3 + MM);
    k_knn<<<g(NN), blk256, 0, stream>>>(pos, fps, e1, ek1);
    k_scatter<<<g((long)NN * CIN), blk256, 0, stream>>>(x, pos, e1, ek1, accX, accP, cnt);
    k_fps_final<<<g(MM * CIN), blk256, 0, stream>>>(accX, accP, cnt, fpsx, fpsp);
    k_kv<<<g(MM * CC), blk256, 0, stream>>>(fpsx, Wk, bk, Wv, bv, xk, xv);

    // ---- f16 conversions ----
    k_cvt_f32_to_f16<<<g((long)NN * CIN), blk256, 0, stream>>>(x, xh, (long)NN * CIN);
    k_cvt_f32_to_f16<<<g(CC * CIN), blk256, 0, stream>>>(Wq, WqH, CC * CIN);
    k_cvt_f32_to_f16<<<g(CC * CC), blk256, 0, stream>>>(pW2, pW2H, CC * CC);
    k_cvt_f32_to_f16<<<g(CC * CC), blk256, 0, stream>>>(aW1, aW1H, CC * CC);
    k_cvt_f32_to_f16<<<g(CC * CC), blk256, 0, stream>>>(aW2, aW2H, CC * CC);

    // ---- x_q = x @ Wq^T + bq  (WMMA, Kd=64) ----
    k_gemm_wmma<<<dim3(NN / 64), dim3(128), 0, stream>>>(xh, WqH, bq, xq, CIN);

    auto bn_stats = [&](const float* gamma, const float* beta) {
        k_zero<<<dim3(1), blk256, 0, stream>>>(stSum, 256);      // zeros sum+sq (contiguous)
        k_stats<<<dim3(256), blk256, 0, stream>>>(Y, stSum, stSq);
        k_bn_finalize<<<dim3(1), dim3(128), 0, stream>>>(stSum, stSq, gamma, beta, stScale, stShift);
    };

    // ---- pos-MLP ----
    k_posmlp1<<<g(ECtot), blk256, 0, stream>>>(pos, fpsp, pW1, pb1, Y);
    bn_stats(pg1, pbe1);
    k_bnrelu_h<<<g(ECtot), blk256, 0, stream>>>(Y, stScale, stShift, AinH);
    k_gemm_wmma<<<dim3(EE / 64), dim3(128), 0, stream>>>(AinH, pW2H, pb2, Y, CC);
    bn_stats(pg2, pbe2);
    k_delta_attnin<<<g(ECtot), blk256, 0, stream>>>(Y, stScale, stShift, xq, xk, delta, AinH);

    // ---- attn-MLP ----
    k_gemm_wmma<<<dim3(EE / 64), dim3(128), 0, stream>>>(AinH, aW1H, ab1, Y, CC);
    bn_stats(ag1, abe1);
    k_bnrelu_h<<<g(ECtot), blk256, 0, stream>>>(Y, stScale, stShift, AinH);
    k_gemm_wmma<<<dim3(EE / 64), dim3(128), 0, stream>>>(AinH, aW2H, ab2, Y, CC);
    bn_stats(ag2, abe2);
    k_bnrelu_inplace<<<g(ECtot), blk256, 0, stream>>>(Y, stScale, stShift);

    // ---- softmax over K + weighted reduction ----
    k_combine<<<g((long)NN * CC), blk256, 0, stream>>>(Y, delta, xv, out);
}
